// TransLayer_7215545057826
// MI455X (gfx1250) — matrix operand order repaired and checked by reading, compiled, not run on hardware
//
#include <hip/hip_runtime.h>
#include <hip/hip_bf16.h>

// ---------------------------------------------------------------------------
// Nystromformer TransLayer for MI455X (gfx1250, wave32, WMMA bf16 path)
// ---------------------------------------------------------------------------
#define TL_DIM    512
#define TL_HEADS  8
#define TL_DHEAD  64
#define TL_N      4096
#define TL_B      4
#define TL_M      256
#define TL_SEGL   16
#define TL_KS     33
#define TL_BH     (TL_B * TL_HEADS)          // 32
#define TL_ROWS   (TL_B * TL_N)              // 16384
#define TL_DSCALE 0.125f                     // DHEAD^-0.5

typedef __attribute__((ext_vector_type(16))) __bf16          bf16x16;
typedef __attribute__((ext_vector_type(16))) unsigned short  u16x16;
typedef __attribute__((ext_vector_type(8)))  unsigned short  u16x8;
typedef __attribute__((ext_vector_type(8)))  float           f32x8;

__device__ __forceinline__ unsigned short f2bf(float f) {
    unsigned int u = __builtin_bit_cast(unsigned int, f);
    u += 0x7FFFu + ((u >> 16) & 1u);   // round-to-nearest-even
    return (unsigned short)(u >> 16);
}

// ---------------------------------------------------------------------------
// Batched bf16 GEMM: C = alpha * A @ op(B) (+bias +resid), templated epilogue
//   A : [M,K] row-major bf16    (M % 64 == 0, K % 64 == 0)
//   B : BT=false -> [K,N] row-major ;  BT=true -> [N,K] row-major (C=A·B^T)
//   NG: enable N bounds guard (only needed when N % 128 != 0; N % 8 == 0 req.)
//   EM: 0 = f32 out | 1 = f32+bf16 out | 2 = bf16 out | 3 = f32 + resid + bias
// Tile: 256 thr = 8 waves; BM=64, BN=128, BK=64; wave tile 32x32 (2x2 WMMA),
// 8 WMMAs per LDS stage; 16B-vector global staging, single wait per stage.
// ---------------------------------------------------------------------------
template <bool BT, bool NG, int EM>
__global__ __launch_bounds__(256)
void gemm_bf16_kernel(const unsigned short* __restrict__ A,
                      const unsigned short* __restrict__ B,
                      float* __restrict__ Cf, unsigned short* __restrict__ Cb,
                      int M, int N, int K,
                      long sA, long sB, long sC,
                      float alpha,
                      const float* __restrict__ resid,
                      const float* __restrict__ bias)
{
    constexpr int BM = 64, BN = 128, BK = 64, PAD = 8;
    constexpr int LDW = BK + PAD;                 // 72 elems -> 144B row stride
    __shared__ unsigned short As[BM][LDW];
    __shared__ unsigned short Bs[BN][LDW];        // N-major: Bs[n][k]

    const int tid  = threadIdx.x;
    const int lane = tid & 31;
    const int wave = tid >> 5;
    const int wr = wave >> 2;          // 0..1  (row wave)
    const int wc = wave & 3;           // 0..3  (col wave)
    const int lm = lane & 15;
    const int lh = lane >> 4;

    const int row0 = blockIdx.y * BM;
    const int col0 = blockIdx.x * BN;
    const long ab = (long)blockIdx.z * sA;
    const long bb = (long)blockIdx.z * sB;
    const long cb = (long)blockIdx.z * sC;

    f32x8 acc[2][2];
#pragma unroll
    for (int r = 0; r < 2; ++r)
#pragma unroll
        for (int c = 0; c < 2; ++c)
#pragma unroll
            for (int e = 0; e < 8; ++e) acc[r][c][e] = 0.0f;

    for (int kk = 0; kk < K; kk += BK) {
        // ---- phase 1: gather tile into registers (16B vector loads) ----
        u16x8 av[2];                               // A: 64x64 = 512 vec8
#pragma unroll
        for (int s = 0; s < 2; ++s) {
            int id = tid + s * 256;
            int r = id >> 3, cv = (id & 7) << 3;
            av[s] = *(const u16x8*)(A + ab + (long)(row0 + r) * K + kk + cv);
        }
        u16x8 bv[4];                               // B: 128x64 = 1024 vec8
#pragma unroll
        for (int s = 0; s < 4; ++s) {
            int id = tid + s * 256;
            if (BT) {
                int n = id >> 3, cv = (id & 7) << 3;
                int gn = col0 + n;
                if (!NG || gn < N)
                    bv[s] = *(const u16x8*)(B + bb + (long)gn * K + kk + cv);
                else
                    bv[s] = (u16x8)0;
            } else {
                int k = id >> 4, nv = (id & 15) << 3;
                int gn = col0 + nv;
                if (!NG || gn < N)
                    bv[s] = *(const u16x8*)(B + bb + (long)(kk + k) * N + gn);
                else
                    bv[s] = (u16x8)0;
            }
        }
        __syncthreads();   // LDS safe to overwrite
        // ---- phase 2: store to LDS ----
#pragma unroll
        for (int s = 0; s < 2; ++s) {
            int id = tid + s * 256;
            int r = id >> 3, cv = (id & 7) << 3;
            *(u16x8*)&As[r][cv] = av[s];
        }
#pragma unroll
        for (int s = 0; s < 4; ++s) {
            int id = tid + s * 256;
            if (BT) {
                int n = id >> 3, cv = (id & 7) << 3;
                *(u16x8*)&Bs[n][cv] = bv[s];
            } else {                                // transpose scatter
                int k = id >> 4, nv = (id & 15) << 3;
#pragma unroll
                for (int j = 0; j < 8; ++j) Bs[nv + j][k] = bv[s][j];
            }
        }
        __syncthreads();

        // ---- phase 3: 2 k-steps x 4 WMMAs from LDS ----
#pragma unroll
        for (int ks = 0; ks < BK / 32; ++ks) {
            const int kb = ks * 32;
            u16x16 ua[2], ub[2];
#pragma unroll
            for (int r = 0; r < 2; ++r) {
                const unsigned short* ap = &As[wr * 32 + r * 16 + lm][kb];
#pragma unroll
                for (int i = 0; i < 16; ++i) {
                    int k = ((i & 8) << 1) + lh * 8 + (i & 7);  // A 16x32 layout
                    ua[r][i] = ap[k];
                }
            }
#pragma unroll
            for (int c = 0; c < 2; ++c) {
                const unsigned short* bp = &Bs[wc * 32 + c * 16 + lm][kb];
#pragma unroll
                for (int i = 0; i < 16; ++i)
                    ub[c][i] = bp[((i & 8) << 1) + lh * 8 + (i & 7)];  // B 32x16 layout
            }
#pragma unroll
            for (int r = 0; r < 2; ++r)
#pragma unroll
                for (int c = 0; c < 2; ++c)
                    acc[r][c] = __builtin_amdgcn_wmma_f32_16x16x32_bf16(
                        false, __builtin_bit_cast(bf16x16, ua[r]),
                        false, __builtin_bit_cast(bf16x16, ub[c]),
                        (short)0, acc[r][c], false, false);
        }
        __syncthreads();
    }

    // ---- epilogue (branch-free, templated): row = v + 8*lh, col = lm ----
#pragma unroll
    for (int r = 0; r < 2; ++r) {
        int gr0 = row0 + wr * 32 + r * 16 + lh * 8;
#pragma unroll
        for (int c = 0; c < 2; ++c) {
            int gc = col0 + wc * 32 + c * 16 + lm;
            if (!NG || gc < N) {
                float rv[8];
                if (EM == 3) {
#pragma unroll
                    for (int v = 0; v < 8; ++v)
                        rv[v] = resid[cb + (long)(gr0 + v) * N + gc];
                }
                float bvv = (EM == 3) ? bias[gc] : 0.0f;
#pragma unroll
                for (int v = 0; v < 8; ++v) {
                    float val = acc[r][c][v] * alpha;
                    if (EM == 3) val += bvv + rv[v];
                    long off = cb + (long)(gr0 + v) * N + gc;
                    if (EM == 0 || EM == 1 || EM == 3) Cf[off] = val;
                    if (EM == 1 || EM == 2)            Cb[off] = f2bf(val);
                }
            }
        }
    }
}

// ---------------------------------------------------------------------------
// f32 -> bf16 buffer conversion (weights)
// ---------------------------------------------------------------------------
__global__ __launch_bounds__(256)
void convert_bf16_kernel(const float* __restrict__ src,
                         unsigned short* __restrict__ dst, long count)
{
    long i = (long)blockIdx.x * 256 + threadIdx.x;
    if (i < count) dst[i] = f2bf(src[i]);
}

// ---------------------------------------------------------------------------
// LayerNorm over DIM=512, block per row, write bf16 directly (GEMM A input)
// ---------------------------------------------------------------------------
__global__ __launch_bounds__(256)
void layernorm_kernel(const float* __restrict__ x,
                      const float* __restrict__ w, const float* __restrict__ b,
                      unsigned short* __restrict__ xnb)
{
    __shared__ float red[256];
    const int t = threadIdx.x;
    const float* xr = x + (long)blockIdx.x * TL_DIM;
    float v0 = xr[t], v1 = xr[t + 256];
    red[t] = v0 + v1;
    __syncthreads();
    for (int o = 128; o > 0; o >>= 1) {
        if (t < o) red[t] += red[t + o];
        __syncthreads();
    }
    float mu = red[0] * (1.0f / TL_DIM);
    __syncthreads();
    float d0 = v0 - mu, d1 = v1 - mu;
    red[t] = d0 * d0 + d1 * d1;
    __syncthreads();
    for (int o = 128; o > 0; o >>= 1) {
        if (t < o) red[t] += red[t + o];
        __syncthreads();
    }
    float rstd = rsqrtf(red[0] * (1.0f / TL_DIM) + 1e-5f);
    unsigned short* out = xnb + (long)blockIdx.x * TL_DIM;
    out[t]       = f2bf(d0 * rstd * w[t]       + b[t]);
    out[t + 256] = f2bf(d1 * rstd * w[t + 256] + b[t + 256]);
}

// ---------------------------------------------------------------------------
// split heads from qkv[16384,1536] -> q_bf (scaled), k_bf, v_bf, v_f32 [bh,n,d]
// ---------------------------------------------------------------------------
__global__ __launch_bounds__(256)
void split_heads_kernel(const float* __restrict__ qkv,
                        unsigned short* __restrict__ qb,
                        unsigned short* __restrict__ kb,
                        unsigned short* __restrict__ vb,
                        float* __restrict__ vf)
{
    long idx = (long)blockIdx.x * 256 + threadIdx.x;   // over B*N*DIM
    int c = (int)(idx % TL_DIM);
    long bn = idx / TL_DIM;                            // b*4096+n
    int n = (int)(bn % TL_N);
    int b = (int)(bn / TL_N);
    int h = c / TL_DHEAD, d = c % TL_DHEAD;
    long src = bn * (3 * TL_DIM) + c;
    long dst = (((long)(b * TL_HEADS + h) * TL_N) + n) * TL_DHEAD + d;
    qb[dst] = f2bf(qkv[src] * TL_DSCALE);
    kb[dst] = f2bf(qkv[src + TL_DIM]);
    float vv = qkv[src + 2 * TL_DIM];
    vb[dst] = f2bf(vv);
    vf[dst] = vv;
}

// ---------------------------------------------------------------------------
// landmarks: q_l,k_l = segment means (l=16) from qkv f32, bf16 out [bh,256,64]
// ---------------------------------------------------------------------------
__global__ __launch_bounds__(64)
void landmark_kernel(const float* __restrict__ qkv,
                     unsigned short* __restrict__ qlb,
                     unsigned short* __restrict__ klb)
{
    int gid = blockIdx.x;                 // bh*256
    int mi = gid & (TL_M - 1);
    int bh = gid >> 8;
    int b = bh / TL_HEADS, h = bh % TL_HEADS;
    int d = threadIdx.x;
    float sq = 0.0f, sk = 0.0f;
    for (int s = 0; s < TL_SEGL; ++s) {
        int n = mi * TL_SEGL + s;
        long base = ((long)(b * TL_N + n)) * (3 * TL_DIM) + h * TL_DHEAD + d;
        sq += qkv[base];
        sk += qkv[base + TL_DIM];
    }
    long o = (long)gid * TL_DHEAD + d;
    qlb[o] = f2bf(sq * (1.0f / TL_SEGL) * TL_DSCALE);
    klb[o] = f2bf(sk * (1.0f / TL_SEGL));
}

// ---------------------------------------------------------------------------
// row softmax (rowlen = PER*256), in-place f32 + bf16 copy
// ---------------------------------------------------------------------------
template <int PER>
__global__ __launch_bounds__(256)
void softmax_kernel(float* __restrict__ X, unsigned short* __restrict__ Xb)
{
    __shared__ float red[256];
    const int t = threadIdx.x;
    long base = (long)blockIdx.x * (PER * 256);
    float vals[PER];
    float mx = -3.4e38f;
#pragma unroll
    for (int i = 0; i < PER; ++i) {
        float v = X[base + t + i * 256];
        vals[i] = v;
        mx = fmaxf(mx, v);
    }
    red[t] = mx;
    __syncthreads();
    for (int o = 128; o > 0; o >>= 1) {
        if (t < o) red[t] = fmaxf(red[t], red[t + o]);
        __syncthreads();
    }
    mx = red[0];
    __syncthreads();
    float sum = 0.0f;
#pragma unroll
    for (int i = 0; i < PER; ++i) {
        vals[i] = __expf(vals[i] - mx);
        sum += vals[i];
    }
    red[t] = sum;
    __syncthreads();
    for (int o = 128; o > 0; o >>= 1) {
        if (t < o) red[t] += red[t + o];
        __syncthreads();
    }
    float inv = 1.0f / red[0];
#pragma unroll
    for (int i = 0; i < PER; ++i) {
        float v = vals[i] * inv;
        X[base + t + i * 256] = v;
        Xb[base + t + i * 256] = f2bf(v);
    }
}

// ---------------------------------------------------------------------------
// pinv norm reductions
// ---------------------------------------------------------------------------
__global__ __launch_bounds__(256)
void rowsum_abs_kernel(const float* __restrict__ a, float* __restrict__ out)
{
    __shared__ float red[256];
    const int t = threadIdx.x;
    red[t] = fabsf(a[(long)blockIdx.x * TL_M + t]);
    __syncthreads();
    for (int o = 128; o > 0; o >>= 1) {
        if (t < o) red[t] += red[t + o];
        __syncthreads();
    }
    if (t == 0) out[blockIdx.x] = red[0];
}

__global__ __launch_bounds__(256)
void colsum_abs_kernel(const float* __restrict__ a, float* __restrict__ out)
{
    __shared__ float red[256];
    const int t = threadIdx.x;
    int bh = blockIdx.x >> 8, col = blockIdx.x & (TL_M - 1);
    red[t] = fabsf(a[(long)bh * TL_M * TL_M + (long)t * TL_M + col]);
    __syncthreads();
    for (int o = 128; o > 0; o >>= 1) {
        if (t < o) red[t] += red[t + o];
        __syncthreads();
    }
    if (t == 0) out[blockIdx.x] = red[0];
}

__global__ __launch_bounds__(256)
void pinv_scale_kernel(const float* __restrict__ rows,
                       const float* __restrict__ cols,
                       float* __restrict__ scale)
{
    __shared__ float r1[256], r2[256];
    const int t = threadIdx.x;
    float mr = 0.0f, mc = 0.0f;
    for (int i = t; i < TL_BH * TL_M; i += 256) {
        mr = fmaxf(mr, rows[i]);
        mc = fmaxf(mc, cols[i]);
    }
    r1[t] = mr; r2[t] = mc;
    __syncthreads();
    for (int o = 128; o > 0; o >>= 1) {
        if (t < o) {
            r1[t] = fmaxf(r1[t], r1[t + o]);
            r2[t] = fmaxf(r2[t], r2[t + o]);
        }
        __syncthreads();
    }
    if (t == 0) scale[0] = 1.0f / (r1[0] * r2[0]);
}

// z0 = a2^T * scale  (bf16)
__global__ __launch_bounds__(256)
void pinv_init_kernel(const float* __restrict__ a2, const float* __restrict__ scale,
                      unsigned short* __restrict__ zb)
{
    long idx = (long)blockIdx.x * 256 + threadIdx.x;   // bh*256*256
    int j = (int)(idx & (TL_M - 1));
    long r = idx >> 8;
    int i = (int)(r & (TL_M - 1));
    long bh = r >> 8;
    zb[idx] = f2bf(a2[bh * TL_M * TL_M + (long)j * TL_M + i] * scale[0]);
}

// W = alpha*I - X  (bf16)
__global__ __launch_bounds__(256)
void diag_sub_kernel(float alpha, const float* __restrict__ X,
                     unsigned short* __restrict__ W)
{
    long idx = (long)blockIdx.x * 256 + threadIdx.x;
    int j = (int)(idx & (TL_M - 1));
    int i = (int)((idx >> 8) & (TL_M - 1));
    float v = -X[idx];
    if (i == j) v += alpha;
    W[idx] = f2bf(v);
}

// ---------------------------------------------------------------------------
// out_attn[bh,n,d] + depthwise conv(v) -> merged bf16 [b,n,h*d]
// ---------------------------------------------------------------------------
__global__ __launch_bounds__(256)
void conv_merge_kernel(const float* __restrict__ oat,
                       const float* __restrict__ vf,
                       const float* __restrict__ cw,
                       unsigned short* __restrict__ merged)
{
    long idx = (long)blockIdx.x * 256 + threadIdx.x;   // bh*n*d
    int d = (int)(idx % TL_DHEAD);
    long r = idx / TL_DHEAD;
    int n = (int)(r % TL_N);
    long bh = r / TL_N;
    int h = (int)(bh % TL_HEADS);
    int b = (int)(bh / TL_HEADS);
    float acc = oat[idx];
    const float* vbase = vf + bh * (long)TL_N * TL_DHEAD + d;
#pragma unroll
    for (int t = 0; t < TL_KS; ++t) {
        int nn = n + t - TL_KS / 2;
        if (nn >= 0 && nn < TL_N)
            acc += vbase[(long)nn * TL_DHEAD] * cw[h * TL_KS + t];
    }
    merged[((long)(b * TL_N + n)) * TL_DIM + h * TL_DHEAD + d] = f2bf(acc);
}

// ---------------------------------------------------------------------------
// host-side orchestration
// ---------------------------------------------------------------------------
extern "C" void kernel_launch(void* const* d_in, const int* in_sizes, int n_in,
                              void* d_out, int out_size, void* d_ws, size_t ws_size,
                              hipStream_t stream)
{
    const float* x      = (const float*)d_in[0];
    const float* norm_w = (const float*)d_in[1];
    const float* norm_b = (const float*)d_in[2];
    const float* Wqkv   = (const float*)d_in[3];
    const float* Wout   = (const float*)d_in[4];
    const float* bout   = (const float*)d_in[5];
    const float* convw  = (const float*)d_in[6];
    float* out = (float*)d_out;

    char* p = (char*)d_ws;
    auto alloc = [&](size_t bytes) -> char* {
        char* r = p;
        p += (bytes + 255) & ~((size_t)255);
        return r;
    };

    const long HND = (long)TL_BH * TL_N * TL_DHEAD;       // 8.4M
    const long MM  = (long)TL_BH * TL_M * TL_M;           // 2.1M
    const long NM  = (long)TL_BH * TL_N * TL_M;           // 33.5M

    unsigned short* xnb   = (unsigned short*)alloc((long)TL_ROWS * TL_DIM * 2);
    unsigned short* wqkvb = (unsigned short*)alloc((long)TL_DIM * 3 * TL_DIM * 2);
    unsigned short* woutb = (unsigned short*)alloc((long)TL_DIM * TL_DIM * 2);
    float*          qkvf  = (float*)alloc((long)TL_ROWS * 3 * TL_DIM * 4);
    unsigned short* qb    = (unsigned short*)alloc(HND * 2);
    unsigned short* kb    = (unsigned short*)alloc(HND * 2);
    unsigned short* vb    = (unsigned short*)alloc(HND * 2);
    float*          vf    = (float*)alloc(HND * 4);
    unsigned short* qlb   = (unsigned short*)alloc((long)TL_BH * TL_M * TL_DHEAD * 2);
    unsigned short* klb   = (unsigned short*)alloc((long)TL_BH * TL_M * TL_DHEAD * 2);
    float*          a1f   = (float*)alloc(NM * 4);        // reused for t_bf
    unsigned short* a1b   = (unsigned short*)alloc(NM * 2);
    float*          a2f   = (float*)alloc(MM * 4);
    unsigned short* a2b   = (unsigned short*)alloc(MM * 2);
    float*          a3f   = (float*)alloc(NM * 4);        // reused for out_attn
    unsigned short* a3b   = (unsigned short*)alloc(NM * 2);
    unsigned short* zb0   = (unsigned short*)alloc(MM * 2);
    unsigned short* zb1   = (unsigned short*)alloc(MM * 2);
    float*          azf   = (float*)alloc(MM * 4);
    unsigned short* azb   = (unsigned short*)alloc(MM * 2);
    float*          wf    = (float*)alloc(MM * 4);
    unsigned short* wb    = (unsigned short*)alloc(MM * 2);
    unsigned short* sb    = (unsigned short*)alloc((long)TL_BH * TL_M * TL_DHEAD * 2);
    float*          rows  = (float*)alloc((long)TL_BH * TL_M * 4);
    float*          cols  = (float*)alloc((long)TL_BH * TL_M * 4);
    float*          scl   = (float*)alloc(256);
    // buffer reuse
    unsigned short* tb      = (unsigned short*)a1f;   // [bh,4096,256] bf16
    float*          oattn   = a3f;                    // [bh,4096,64]  f32
    unsigned short* mergedb = (unsigned short*)qkvf;  // [16384,512]   bf16
    (void)in_sizes; (void)n_in; (void)out_size; (void)ws_size;

    // 1) weight conversion + layernorm
    convert_bf16_kernel<<<(TL_DIM * 3 * TL_DIM) / 256, 256, 0, stream>>>(Wqkv, wqkvb, (long)TL_DIM * 3 * TL_DIM);
    convert_bf16_kernel<<<(TL_DIM * TL_DIM) / 256, 256, 0, stream>>>(Wout, woutb, (long)TL_DIM * TL_DIM);
    layernorm_kernel<<<TL_ROWS, 256, 0, stream>>>(x, norm_w, norm_b, xnb);

    // 2) qkv = xn @ Wqkv   [16384 x 512 x 1536]
    gemm_bf16_kernel<false, false, 0><<<dim3(12, 256, 1), 256, 0, stream>>>(
        xnb, wqkvb, qkvf, nullptr, TL_ROWS, 3 * TL_DIM, TL_DIM, 0, 0, 0, 1.0f, nullptr, nullptr);

    // 3) head split + landmarks
    split_heads_kernel<<<(unsigned)(((long)TL_ROWS * TL_DIM) / 256), 256, 0, stream>>>(qkvf, qb, kb, vb, vf);
    landmark_kernel<<<TL_BH * TL_M, 64, 0, stream>>>(qkvf, qlb, klb);

    const long sQ = (long)TL_N * TL_DHEAD, sL = (long)TL_M * TL_DHEAD;
    const long sNM = (long)TL_N * TL_M, sMM = (long)TL_M * TL_M;
    const long sMN = (long)TL_M * TL_N, sND = (long)TL_N * TL_DHEAD;

    // 4) sim1 = q @ k_l^T -> softmax -> a1
    gemm_bf16_kernel<true, false, 0><<<dim3(2, 64, TL_BH), 256, 0, stream>>>(
        qb, klb, a1f, nullptr, TL_N, TL_M, TL_DHEAD, sQ, sL, sNM, 1.0f, nullptr, nullptr);
    softmax_kernel<1><<<TL_BH * TL_N, 256, 0, stream>>>(a1f, a1b);

    // 5) sim2 = q_l @ k_l^T -> softmax -> a2
    gemm_bf16_kernel<true, false, 0><<<dim3(2, 4, TL_BH), 256, 0, stream>>>(
        qlb, klb, a2f, nullptr, TL_M, TL_M, TL_DHEAD, sL, sL, sMM, 1.0f, nullptr, nullptr);
    softmax_kernel<1><<<TL_BH * TL_M, 256, 0, stream>>>(a2f, a2b);

    // 6) sim3 = q_l @ k^T -> softmax -> a3
    gemm_bf16_kernel<true, false, 0><<<dim3(32, 4, TL_BH), 256, 0, stream>>>(
        qlb, kb, a3f, nullptr, TL_M, TL_N, TL_DHEAD, sL, sQ, sMN, 1.0f, nullptr, nullptr);
    softmax_kernel<16><<<TL_BH * TL_M, 256, 0, stream>>>(a3f, a3b);

    // 7) pinv init: z = a2^T / (max row-sum * max col-sum)
    rowsum_abs_kernel<<<TL_BH * TL_M, 256, 0, stream>>>(a2f, rows);
    colsum_abs_kernel<<<TL_BH * TL_M, 256, 0, stream>>>(a2f, cols);
    pinv_scale_kernel<<<1, 256, 0, stream>>>(rows, cols, scl);
    pinv_init_kernel<<<(unsigned)(MM / 256), 256, 0, stream>>>(a2f, scl, zb0);

    // 8) 6 Newton-Schulz iterations: z = 0.25 z (13I - az(15I - az(7I - az)))
    unsigned short* zc = zb0;
    unsigned short* zn = zb1;
    const dim3 gmm(2, 4, TL_BH);
    const unsigned dge = (unsigned)(MM / 256);
    for (int it = 0; it < 6; ++it) {
        gemm_bf16_kernel<false, false, 1><<<gmm, 256, 0, stream>>>(
            a2b, zc, azf, azb, TL_M, TL_M, TL_M, sMM, sMM, sMM, 1.0f, nullptr, nullptr);
        diag_sub_kernel<<<dge, 256, 0, stream>>>(7.0f, azf, wb);
        gemm_bf16_kernel<false, false, 0><<<gmm, 256, 0, stream>>>(
            azb, wb, wf, nullptr, TL_M, TL_M, TL_M, sMM, sMM, sMM, 1.0f, nullptr, nullptr);
        diag_sub_kernel<<<dge, 256, 0, stream>>>(15.0f, wf, wb);
        gemm_bf16_kernel<false, false, 0><<<gmm, 256, 0, stream>>>(
            azb, wb, wf, nullptr, TL_M, TL_M, TL_M, sMM, sMM, sMM, 1.0f, nullptr, nullptr);
        diag_sub_kernel<<<dge, 256, 0, stream>>>(13.0f, wf, wb);
        gemm_bf16_kernel<false, false, 2><<<gmm, 256, 0, stream>>>(
            zc, wb, nullptr, zn, TL_M, TL_M, TL_M, sMM, sMM, sMM, 0.25f, nullptr, nullptr);
        unsigned short* tmp = zc; zc = zn; zn = tmp;
    }

    // 9) t = a1 @ z ; s = a3 @ v ; out_attn = t @ s
    gemm_bf16_kernel<false, false, 2><<<dim3(2, 64, TL_BH), 256, 0, stream>>>(
        a1b, zc, nullptr, tb, TL_N, TL_M, TL_M, sNM, sMM, sNM, 1.0f, nullptr, nullptr);
    gemm_bf16_kernel<false, true, 2><<<dim3(1, 4, TL_BH), 256, 0, stream>>>(
        a3b, vb, nullptr, sb, TL_M, TL_DHEAD, TL_N, sMN, sND, sL, 1.0f, nullptr, nullptr);
    gemm_bf16_kernel<false, true, 0><<<dim3(1, 64, TL_BH), 256, 0, stream>>>(
        tb, sb, oattn, nullptr, TL_N, TL_DHEAD, TL_M, sNM, sL, sND, 1.0f, nullptr, nullptr);

    // 10) depthwise conv residual + merge heads -> bf16 [16384,512]
    conv_merge_kernel<<<(unsigned)(HND / 256), 256, 0, stream>>>(oattn, vf, convw, mergedb);

    // 11) final: out = x + merged @ Wout + bout
    gemm_bf16_kernel<false, false, 3><<<dim3(4, 256, 1), 256, 0, stream>>>(
        mergedb, woutb, out, nullptr, TL_ROWS, TL_DIM, TL_DIM, 0, 0, 0, 1.0f, x, bout);
}
